// Trans_7842610283295
// MI455X (gfx1250) — matrix operand order repaired and checked by reading, not compiled, not run on hardware
//
#include <hip/hip_runtime.h>
#include <math.h>

#define CHN   64
#define GRP   4
#define BATCH 2
#define Hdim  256
#define Wdim  256
#define HP    258            /* padded spatial (1-halo each side) */
#define HW    (Hdim*Wdim)
#define NCHW  ((size_t)BATCH*CHN*HW)
#define MAXP  16384          /* worst-case patches per (b,g): (256/2)*(256/2) */
#define STATN ((size_t)BATCH*CHN*MAXP)
#define EPSV  1e-9f
#define ALPHAV 0.9f

#define W1SZ (4*9*80*32)     /* conv1+sw prepped B tiles: [cb][tap][n=80][k=32] */
#define WGSZ (4*5*16*32)     /* grouped conv prepped B tiles: [g][t][n=16][k=32] */
#define CL1SZ ((size_t)BATCH*HP*HP*128)   /* padded NHWC f16 concat input */
#define CLXSZ ((size_t)BATCH*HP*HP*64)    /* padded NHWC f16 x1 / x2 */

typedef __attribute__((ext_vector_type(16))) _Float16 v16h;
typedef __attribute__((ext_vector_type(8)))  _Float16 v8h;
typedef __attribute__((ext_vector_type(8)))  float    v8f;

union AF { v16h v; _Float16 h[16]; struct { v8h lo, hi; } p; };
union CF { v8f  v; float    f[8];  };
union H8 { v8h  v; _Float16 h[8];  };

__device__ __forceinline__ v8f wmma16x16x32(v16h a, v16h b, v8f c){
  return __builtin_amdgcn_wmma_f32_16x16x32_f16(false, a, false, b, (short)0, c, false, false);
}

// Async Global->LDS 16B copy (gfx1250, GVS mode): tracked by ASYNCcnt.
__device__ __forceinline__ void async_ld_b128(const void* gbase, unsigned goff, void* lptr){
  unsigned loff = (unsigned)(size_t)lptr;   // low 32 bits of generic LDS ptr = LDS byte offset
  asm volatile("global_load_async_to_lds_b128 %0, %1, %2"
               :: "v"(loff), "v"(goff), "s"((unsigned long long)(size_t)gbase)
               : "memory");
}
__device__ __forceinline__ void wait_async0(){
  asm volatile("s_wait_asynccnt 0x0" ::: "memory");
}

__device__ __forceinline__ float mishf(float x){
  float sp = (x > 20.f) ? x : log1pf(expf(x));
  return x * tanhf(sp);
}
__device__ __forceinline__ float sigmoidf(float x){ return 1.f/(1.f+expf(-x)); }

// ---------------------------------------------------------------- init
__global__ void init_kernel(float* p_acc){
  int t = threadIdx.x;
  if (t < BATCH*16) p_acc[t] = 0.f;
}

// ------------------- prep f16 B-fragment weight tiles (K-contiguous per n)
__global__ void prep_w1_kernel(const float* cw1, const float* sw, _Float16* wB1){
  int i = blockIdx.x*256 + threadIdx.x;
  if (i >= W1SZ) return;
  int k   = i & 31;
  int n   = (i >> 5) % 80;
  int tap = (i / (32*80)) % 9;
  int cb  = i / (32*80*9);
  int c   = cb*32 + k;
  float v = (n < 64) ? cw1[((size_t)n*128 + c)*9 + tap]
                     : sw [((size_t)(n-64)*128 + c)*9 + tap];
  wB1[i] = (_Float16)v;
}

__global__ void prep_wg_kernel(const float* w, _Float16* wB){
  int i = blockIdx.x*256 + threadIdx.x;
  if (i >= WGSZ) return;
  int k = i & 31;
  int n = (i >> 5) & 15;
  int t = (i >> 9) % 5;
  int g = (i >> 9) / 5;
  int tap = (k < 16) ? 2*t : 2*t + 1;      // tap pair packing; tap==9 -> zero
  int ch  = k & 15;
  float v = (tap < 9) ? w[((size_t)(g*16+n)*16 + ch)*9 + tap] : 0.f;
  wB[i] = (_Float16)v;
}

// ------------- prep padded NHWC f16 concat input [b][hp][wp][128], zero halo
__global__ void prep_cl1_kernel(const float* vis, const float* irp, _Float16* cl1){
  size_t i = (size_t)blockIdx.x*256 + threadIdx.x;            // one per 8-ch group
  if (i >= (size_t)BATCH*HP*HP*16) return;
  int k8 = (int)(i & 15);
  size_t rest = i >> 4;
  int wp = (int)(rest % HP);
  int hp = (int)((rest / HP) % HP);
  int b  = (int)(rest / ((size_t)HP*HP));
  bool inter = (hp >= 1 && hp <= Hdim && wp >= 1 && wp <= Wdim);
  H8 t;
  #pragma unroll
  for (int j = 0; j < 8; j++){
    int c = k8*8 + j;
    float v = 0.f;
    if (inter){
      const float* src = (c < CHN) ? vis : irp;
      int cc = (c < CHN) ? c : c - CHN;
      v = src[(((size_t)b*CHN + cc)*Hdim + (hp-1))*Wdim + (wp-1)];
    }
    t.h[j] = (_Float16)v;
  }
  *reinterpret_cast<v8h*>(&cl1[(((size_t)b*HP + hp)*HP + wp)*128 + k8*8]) = t.v;
}

// --------------------- zero the halo of a padded NHWC f16 64-ch buffer
__global__ void zero_halo_kernel(_Float16* buf){
  size_t i = (size_t)blockIdx.x*256 + threadIdx.x;            // one per 8-ch group
  if (i >= (size_t)BATCH*HP*HP*8) return;
  int k8 = (int)(i & 7);
  size_t rest = i >> 3;
  int wp = (int)(rest % HP);
  int hp = (int)((rest / HP) % HP);
  int b  = (int)(rest / ((size_t)HP*HP));
  if (hp >= 1 && hp <= Hdim && wp >= 1 && wp <= Wdim) return;  // interior: conv writes it
  H8 t;
  #pragma unroll
  for (int j = 0; j < 8; j++) t.h[j] = (_Float16)0.f;
  *reinterpret_cast<v8h*>(&buf[(((size_t)b*HP + hp)*HP + wp)*64 + k8*8]) = t.v;
}

// ------------------------------------------------- channel means (a, bb)
__global__ void mean_kernel(const float* ir_feat, const float* vis_feat,
                            float* amean, float* bmean){
  int c = blockIdx.x, b = blockIdx.y, sel = blockIdx.z;
  const float* src  = (sel==0)? ir_feat : vis_feat;
  const float* base = src + ((size_t)b*CHN + c)*HW;
  float s = 0.f;
  for (int i = threadIdx.x; i < HW; i += 256) s += base[i];
  __shared__ float red[256];
  red[threadIdx.x] = s; __syncthreads();
  for (int st = 128; st > 0; st >>= 1){
    if (threadIdx.x < st) red[threadIdx.x] += red[threadIdx.x+st];
    __syncthreads();
  }
  if (threadIdx.x == 0){
    float m = red[0] / (float)HW;
    if (sel==0) amean[b*CHN+c] = m; else bmean[b*CHN+c] = m;
  }
}

// ---------------------------------------------------- intensity MLP head
__global__ void mlp_kernel(const float* amean, const float* bmean,
  const float* iw1, const float* ib1, const float* iw2, const float* ib2,
  const float* iw3, const float* ib3, const float* iw4, const float* ib4,
  float* inten)
{
  __shared__ float shf[128];
  __shared__ float shf3[64];
  int j = threadIdx.x;
  for (int b = 0; b < BATCH; b++){
    float s1 = ib1[j], s2 = ib2[j];
    for (int k = 0; k < 64; k++){
      s1 += amean[b*64+k]*iw1[j*64+k];
      s2 += bmean[b*64+k]*iw2[j*64+k];
    }
    shf[j] = mishf(s1); shf[64+j] = mishf(s2);
    __syncthreads();
    float s3 = ib3[j];
    for (int k = 0; k < 128; k++) s3 += shf[k]*iw3[j*128+k];
    shf3[j] = mishf(s3);
    __syncthreads();
    float s4 = ib4[j];
    for (int k = 0; k < 64; k++) s4 += shf3[k]*iw4[j*64+k];
    inten[b*64+j] = sigmoidf(s4);
    __syncthreads();
  }
}

// ------------------------------------- fused conv1 (128->64) + sw (128->16)
// Implicit GEMM, WMMA f32<-f16, async Global->LDS staging from the padded
// NHWC f16 input. 8 waves x 16-pixel M strips x 5 N-tiles; K = 4 cin blocks
// of 32 x 9 taps.
__global__ __launch_bounds__(256) void conv1_kernel(
  const _Float16* cl1, const _Float16* wB1,
  const float* cg1, const float* cb1, const float* sg, const float* sb,
  _Float16* xA, float* p_acc)
{
  __shared__ __align__(16) _Float16 lin[3*130*32];  // (3 rows)(128+2 halo)(32 ch)
  const int tid  = threadIdx.x;
  const int lane = tid & 31;
  const int wv   = tid >> 5;
  const int b    = blockIdx.z;
  const int h    = blockIdx.y;
  const int wb0  = blockIdx.x * 128;
  const int m    = lane & 15;
  const bool hi  = lane >= 16;
  const int kbase = hi ? 8 : 0;

  CF acc[5];
  #pragma unroll
  for (int nt = 0; nt < 5; nt++)
    #pragma unroll
    for (int r = 0; r < 8; r++) acc[nt].f[r] = 0.f;

  for (int cb = 0; cb < 4; cb++){
    __syncthreads();
    // async-stage 1560 x 16B chunks: (row, col, q) -> 8 halves of channel run
    for (int idx = tid; idx < 3*130*4; idx += 256){
      int q = idx & 3, rest = idx >> 2;
      int col = rest % 130, row = rest / 130;
      unsigned goff = (unsigned)(((((size_t)b*HP + (h+row))*HP + (wb0+col))*128
                                  + cb*32 + q*8) * 2);
      async_ld_b128(cl1, goff, &lin[(size_t)(row*130 + col)*32 + q*8]);
    }
    wait_async0();
    __syncthreads();

    const _Float16* wb = wB1 + (size_t)cb*(9*80*32);
    #pragma unroll
    for (int tap = 0; tap < 9; tap++){
      const int dy = tap/3 - 1, dx = tap%3 - 1;
      const int colb = wv*16 + m + dx + 1;
      const int offA = ((dy+1)*130 + colb)*32;
      AF a;   // A 16x32 f16 layout: runs [base,base+8) and [base+16,base+24)
      a.p.lo = *reinterpret_cast<const v8h*>(&lin[offA + kbase]);
      a.p.hi = *reinterpret_cast<const v8h*>(&lin[offA + kbase + 16]);
      const _Float16* wt = wb + (size_t)tap*(80*32) + (size_t)m*32 + (hi ? 8 : 0);
      #pragma unroll
      for (int nt = 0; nt < 5; nt++){
        AF bf;  // B KxN layout: K runs [base,base+8) and [base+16,base+24), base = hi?8:0
        const _Float16* p = wt + (size_t)nt*(16*32);
        bf.p.lo = *reinterpret_cast<const v8h*>(p);
        bf.p.hi = *reinterpret_cast<const v8h*>(p + 16);
        acc[nt].v = wmma16x16x32(a.v, bf.v, acc[nt].v);
      }
    }
  }
  // epilogue: conv1 -> CBR -> padded NHWC f16 xA ; sw -> CBR -> p partial sums
  #pragma unroll
  for (int nt = 0; nt < 4; nt++){
    int n = nt*16 + m;
    float g = cg1[n], be = cb1[n];
    #pragma unroll
    for (int r = 0; r < 8; r++){
      int mp = r + (hi ? 8 : 0);
      int pix = wb0 + wv*16 + mp;
      float val = fmaxf(acc[nt].f[r]*g + be, 0.f);
      xA[(((size_t)b*HP + (h+1))*HP + (pix+1))*64 + n] = (_Float16)val;
    }
  }
  {
    int n = m;
    float g = sg[n], be = sb[n];
    float s = 0.f;
    #pragma unroll
    for (int r = 0; r < 8; r++) s += fmaxf(acc[4].f[r]*g + be, 0.f);
    atomicAdd(&p_acc[b*16 + n], s);
  }
}

// ------------------------------------------ scale/ratio -> patch geometry
__global__ void params_kernel(const float* p_acc, const float* ssw, const float* ssb,
                              const float* srw, const float* srb, int* params){
  int t = threadIdx.x;
  if (t >= BATCH*GRP) return;
  int b = t / GRP, g = t % GRP;
  const int PL[4] = {8, 16, 32, 32};
  float s = ssb[g], r = srb[g];
  for (int k = 0; k < 16; k++){
    float pm = p_acc[b*16+k] / (float)HW;
    s += pm * ssw[g*16+k];
    r += pm * srw[g*16+k];
  }
  float scale = ALPHAV*tanhf(s) + 1.f;
  float ratio = ALPHAV*tanhf(r) + 1.f;
  int pw0 = (int)rintf(scale * (float)PL[g]);   // rintf == Python round (half-even)
  int ph0 = (int)rintf(ratio * (float)pw0);     // uses UNCLAMPED pw0 (per reference)
  int pw = min(max(2, pw0), Wdim);
  int ph = min(max(2, ph0), Hdim);
  int nh = (Hdim + ph - 1)/ph;
  int nw = (Wdim + pw - 1)/pw;
  params[t*4+0]=ph; params[t*4+1]=pw; params[t*4+2]=nh; params[t*4+3]=nw;
}

// ---------------------------- grouped 3x3 conv (4 x 16->16), WMMA, 2 taps/K32
// Input: padded NHWC f16 -> async contiguous row copies into LDS.
// mode 0: relu(acc*p0+p1) -> padded NHWC f16 outH ; mode 1: sigmoid(acc+p0)
// -> NHWC f32 outF (conf).
__global__ __launch_bounds__(256) void conv_group_kernel(
  const _Float16* xin, const _Float16* wB, const float* p0, const float* p1,
  _Float16* outH, float* outF, int mode)
{
  __shared__ __align__(16) _Float16 lin[3*34*64];  // (3 rows)(32+2 halo)(64 ch)
  const int tid = threadIdx.x, lane = tid & 31, wv = tid >> 5;
  const int b = blockIdx.z, h = blockIdx.y, wb0 = blockIdx.x*32;
  const int g = wv & 3, mt = wv >> 2;          // 8 waves = 4 groups x 2 M-tiles
  const int m = lane & 15; const bool hi = lane >= 16;
  const int kbase = hi ? 8 : 0;

  v8h zero8;
  #pragma unroll
  for (int j = 0; j < 8; j++) zero8[j] = (_Float16)0.f;

  // async-stage 3 contiguous row segments of 34*64 halves (4352 B) each
  for (int idx = tid; idx < 3*272; idx += 256){
    int row = idx / 272, inb = (idx % 272) * 16;   // byte offset within row seg
    unsigned goff = (unsigned)(((((size_t)b*HP + (h+row))*HP + wb0)*64)*2 + inb);
    async_ld_b128(xin, goff, (char*)lin + row*(34*64*2) + inb);
  }
  wait_async0();
  __syncthreads();

  CF acc;
  #pragma unroll
  for (int r = 0; r < 8; r++) acc.f[r] = 0.f;

  #pragma unroll
  for (int t = 0; t < 5; t++){
    const int tap0 = 2*t, tap1 = 2*t + 1;      // tap1==9 on last iter -> zeros
    AF a;
    {
      int dy = tap0/3 - 1, dx = tap0%3 - 1;
      int off = ((dy+1)*34 + (mt*16 + m + dx + 1))*64 + g*16 + kbase;
      a.p.lo = *reinterpret_cast<const v8h*>(&lin[off]);
    }
    if (tap1 < 9){
      int dy = tap1/3 - 1, dx = tap1%3 - 1;
      int off = ((dy+1)*34 + (mt*16 + m + dx + 1))*64 + g*16 + kbase;
      a.p.hi = *reinterpret_cast<const v8h*>(&lin[off]);
    } else {
      a.p.hi = zero8;
    }
    AF bf;
    const _Float16* p = wB + ((size_t)((g*5 + t)*16 + m))*32 + (hi ? 8 : 0);
    bf.p.lo = *reinterpret_cast<const v8h*>(p);
    bf.p.hi = *reinterpret_cast<const v8h*>(p + 16);
    acc.v = wmma16x16x32(a.v, bf.v, acc.v);
  }

  {
    int oc = g*16 + m;
    float a0 = p0[oc];
    float a1 = (mode == 0) ? p1[oc] : 0.f;
    #pragma unroll
    for (int r = 0; r < 8; r++){
      int mp = r + (hi ? 8 : 0);
      int pix = wb0 + mt*16 + mp;
      float val = acc.f[r];
      if (mode == 0){
        val = fmaxf(val*a0 + a1, 0.f);
        outH[(((size_t)b*HP + (h+1))*HP + (pix+1))*64 + oc] = (_Float16)val;
      } else {
        val = sigmoidf(val + a0);
        outF[(((size_t)b*Hdim + h)*Wdim + pix)*64 + oc] = val;
      }
    }
  }
}

// ------------------------------- per-patch mean/std (reflect pad), irf & vis
__global__ void stats_kernel(const float* ir_feat, const float* vis_feat,
  const float* inten, const int* params,
  float* irm, float* irs, float* vm, float* vs)
{
  int pidx = blockIdx.x, c = blockIdx.y, b = blockIdx.z;
  int g = c >> 4;
  const int* pp = params + (b*GRP + g)*4;
  int ph = pp[0], pw = pp[1], nh = pp[2], nw = pp[3];
  if (pidx >= nh*nw) return;
  int iy = pidx / nw, ix = pidx % nw;
  float it = inten[b*CHN + c];
  const float* irb = ir_feat  + ((size_t)b*CHN + c)*HW;
  const float* vb  = vis_feat + ((size_t)b*CHN + c)*HW;
  float s0=0,s1=0,s2=0,s3=0;
  int np = ph*pw;
  for (int e = threadIdx.x; e < np; e += 128){
    int row = iy*ph + e/pw;
    int col = ix*pw + e%pw;
    if (row >= Hdim) row = 2*Hdim - 2 - row;   // jnp.pad mode='reflect'
    if (col >= Wdim) col = 2*Wdim - 2 - col;
    float xi = it * irb[row*Wdim + col];
    float xv = vb[row*Wdim + col];
    s0 += xi; s1 += xi*xi; s2 += xv; s3 += xv*xv;
  }
  __shared__ float red[4*128];
  int tid = threadIdx.x;
  red[tid]=s0; red[128+tid]=s1; red[256+tid]=s2; red[384+tid]=s3;
  __syncthreads();
  for (int st = 64; st > 0; st >>= 1){
    if (tid < st){
      red[tid]+=red[tid+st]; red[128+tid]+=red[128+tid+st];
      red[256+tid]+=red[256+tid+st]; red[384+tid]+=red[384+tid+st];
    }
    __syncthreads();
  }
  if (tid == 0){
    float inv = 1.f/(float)np;
    float m0 = red[0]*inv,   m2 = red[256]*inv;
    float v0 = fmaxf(red[128]*inv - m0*m0, 0.f);
    float v2 = fmaxf(red[384]*inv - m2*m2, 0.f);
    size_t o = ((size_t)b*CHN + c)*MAXP + pidx;
    irm[o]=m0; irs[o]=sqrtf(v0); vm[o]=m2; vs[o]=sqrtf(v2);
  }
}

// -------------------------------------------------- final transfer + blend
__global__ void transfer_kernel(const float* ir_feat, const float* vis_feat,
  const float* conf, const float* inten, const int* params,
  const float* irm, const float* irs, const float* vm, const float* vs,
  float* out)
{
  size_t i = (size_t)blockIdx.x*blockDim.x + threadIdx.x;
  if (i >= NCHW) return;
  int wq = (int)(i % Wdim);
  int hq = (int)((i / Wdim) % Hdim);
  int c  = (int)((i / HW) % CHN);
  int b  = (int)(i / ((size_t)CHN*HW));
  int g = c >> 4;
  const int* pp = params + (b*GRP + g)*4;
  int ph = pp[0], pw = pp[1], nw = pp[3];
  int pidx = (hq/ph)*nw + (wq/pw);
  size_t so = ((size_t)b*CHN + c)*MAXP + pidx;
  float xi = inten[b*CHN + c] * ir_feat[i];
  float xv = vis_feat[i];
  float tr = (xi - irm[so]) / (irs[so] + EPSV) * vs[so] + vm[so];
  float cf = conf[(((size_t)b*Hdim + hq)*Wdim + wq)*64 + c];   // NHWC conf
  out[i] = cf*tr + (1.f - cf)*xv;
}

// -----------------------------------------------------------------------
extern "C" void kernel_launch(void* const* d_in, const int* in_sizes, int n_in,
                              void* d_out, int out_size, void* d_ws, size_t ws_size,
                              hipStream_t stream)
{
  (void)in_sizes; (void)n_in; (void)out_size; (void)ws_size;
  const float* vis = (const float*)d_in[0];
  const float* irf = (const float*)d_in[1];
  /* d_in[2] = ir (unused by reference) */
  const float* iw1=(const float*)d_in[3],  *ib1=(const float*)d_in[4];
  const float* iw2=(const float*)d_in[5],  *ib2=(const float*)d_in[6];
  const float* iw3=(const float*)d_in[7],  *ib3=(const float*)d_in[8];
  const float* iw4=(const float*)d_in[9],  *ib4=(const float*)d_in[10];
  const float* cw1=(const float*)d_in[11], *cg1=(const float*)d_in[12], *cb1=(const float*)d_in[13];
  const float* cw2=(const float*)d_in[14], *cg2=(const float*)d_in[15], *cb2=(const float*)d_in[16];
  const float* cw3=(const float*)d_in[17], *cbias3=(const float*)d_in[18];
  const float* sw =(const float*)d_in[19], *sg=(const float*)d_in[20],  *sb=(const float*)d_in[21];
  const float* ssw=(const float*)d_in[22], *ssb=(const float*)d_in[23];
  const float* srw=(const float*)d_in[24], *srb=(const float*)d_in[25];

  // f16 region first (keeps 16B alignment), then f32 region
  _Float16* wB1 = (_Float16*)d_ws;          // W1SZ halves
  _Float16* wB2 = wB1 + W1SZ;
  _Float16* wB3 = wB2 + WGSZ;
  _Float16* cl1 = wB3 + WGSZ;               // padded NHWC input, CL1SZ halves
  _Float16* xA  = cl1 + CL1SZ;              // padded NHWC x1
  _Float16* xB  = xA  + CLXSZ;              // padded NHWC x2
  float* conf   = (float*)(xB + CLXSZ);     // NHWC f32 conf
  float* st_irm = conf + NCHW;
  float* st_irs = st_irm + STATN;
  float* st_vm  = st_irs + STATN;
  float* st_vs  = st_vm + STATN;
  float* amean  = st_vs + STATN;
  float* bmean  = amean + BATCH*CHN;
  float* inten  = bmean + BATCH*CHN;
  float* p_acc  = inten + BATCH*CHN;
  int*   params = (int*)(p_acc + BATCH*16);

  init_kernel<<<1, 64, 0, stream>>>(p_acc);
  prep_w1_kernel<<<(W1SZ + 255)/256, 256, 0, stream>>>(cw1, sw, wB1);
  prep_wg_kernel<<<(WGSZ + 255)/256, 256, 0, stream>>>(cw2, wB2);
  prep_wg_kernel<<<(WGSZ + 255)/256, 256, 0, stream>>>(cw3, wB3);
  {
    size_t n = (size_t)BATCH*HP*HP*16;
    prep_cl1_kernel<<<(unsigned)((n + 255)/256), 256, 0, stream>>>(vis, irf, cl1);
  }
  {
    size_t n = (size_t)BATCH*HP*HP*8;
    zero_halo_kernel<<<(unsigned)((n + 255)/256), 256, 0, stream>>>(xA);
    zero_halo_kernel<<<(unsigned)((n + 255)/256), 256, 0, stream>>>(xB);
  }
  mean_kernel<<<dim3(CHN, BATCH, 2), 256, 0, stream>>>(irf, vis, amean, bmean);
  mlp_kernel<<<1, 64, 0, stream>>>(amean, bmean, iw1,ib1, iw2,ib2, iw3,ib3, iw4,ib4, inten);
  conv1_kernel<<<dim3(Wdim/128, Hdim, BATCH), 256, 0, stream>>>(
      cl1, wB1, cg1, cb1, sg, sb, xA, p_acc);
  params_kernel<<<1, 32, 0, stream>>>(p_acc, ssw, ssb, srw, srb, params);
  conv_group_kernel<<<dim3(Wdim/32, Hdim, BATCH), 256, 0, stream>>>(
      xA, wB2, cg2, cb2, xB, nullptr, 0);
  conv_group_kernel<<<dim3(Wdim/32, Hdim, BATCH), 256, 0, stream>>>(
      xB, wB3, cbias3, nullptr, nullptr, conf, 1);
  stats_kernel<<<dim3(MAXP, CHN, BATCH), 128, 0, stream>>>(
      irf, vis, inten, params, st_irm, st_irs, st_vm, st_vs);
  int blocks = (int)((NCHW + 255)/256);
  transfer_kernel<<<blocks, 256, 0, stream>>>(
      irf, vis, conf, inten, params, st_irm, st_irs, st_vm, st_vs, (float*)d_out);
}
